// SlidingWindowAttention_37864431682613
// MI455X (gfx1250) — matrix operand order repaired and checked by reading, not compiled, not run on hardware
//
#include <hip/hip_runtime.h>
#include <hip/hip_bf16.h>

typedef _Float16 f16;
typedef __attribute__((ext_vector_type(16))) _Float16 v16h;
typedef __attribute__((ext_vector_type(8)))  _Float16 v8h;
typedef __attribute__((ext_vector_type(8)))  float    v8f;
typedef __attribute__((ext_vector_type(4)))  unsigned int u32x4;
typedef __attribute__((ext_vector_type(8)))  int i32x8;
typedef __attribute__((ext_vector_type(4)))  int i32x4;

union V16u { v16h v; v8h h[2]; };

#define T_SEQ   4096
#define C_EMB   1024
#define KV_DIM  256
#define N_HEAD  16
#define N_KVH   4
#define HEAD_D  64
#define WIN     512

#if __has_builtin(__builtin_amdgcn_tensor_load_to_lds)
#define USE_TDM 1
#else
#define USE_TDM 0
#endif

// ---------------------------------------------------------------------------
// WMMA helper: D = A(16x32 f16) * B(32x16 f16) + C(16x16 f32)
// ---------------------------------------------------------------------------
__device__ __forceinline__ v8f wmma_f16(v16h a, v16h b, v8f c) {
  return __builtin_amdgcn_wmma_f32_16x16x32_f16(false, a, false, b, (short)0, c,
                                                false, false);
}

// A-fragment (16x32, 16-bit): lane<16 holds row (lane&15), K {k0..k0+15};
// lane>=16 holds same row, K {k0+16..k0+31}.
__device__ __forceinline__ v16h load_a_frag(const f16* __restrict__ base, int ld,
                                            int row, int k0, int lane) {
  const f16* p = base + (size_t)row * ld + k0 + ((lane >> 4) << 4);
  V16u u;
  u.h[0] = *(const v8h*)p;
  u.h[1] = *(const v8h*)(p + 8);
  return u.v;
}

// ---------------------------------------------------------------------------
// fp32 -> fp16 cast
// ---------------------------------------------------------------------------
__global__ void cast_f32_to_f16(const float* __restrict__ s, f16* __restrict__ d, int n) {
  int i = blockIdx.x * 256 + threadIdx.x;
  if (i < n) d[i] = (f16)s[i];
}

#if USE_TDM
// ---------------------------------------------------------------------------
// Tensor Data Mover: DMA a 64-row x 32-col f16 tile (row stride = ld elements)
// from global into LDS at byte offset lds_off, laid out as [64][32] f16.
// D# built per CDNA5 ISA sect. 8.3/8.4:
//   g0: count=1 | lds_addr | global_addr[56:0] | type=2
//   g1: data_size=1(2B), tensor_dim0=ld, tensor_dim1=nrows,
//       tile_dim0=32, tile_dim1=64, tensor_dim0_stride=ld
// ---------------------------------------------------------------------------
__device__ __forceinline__ void tdm_load_tile64x32(unsigned lds_off,
                                                   const f16* gtile,
                                                   unsigned ld, unsigned nrows) {
  unsigned long long ga = (unsigned long long)gtile;
  u32x4 g0;
  g0.x = 1u;                                   // count=1, user D#, no gather
  g0.y = lds_off;                              // LDS byte address of tile
  g0.z = (unsigned)(ga & 0xffffffffu);         // global_addr[31:0]
  g0.w = ((unsigned)(ga >> 32) & 0x01ffffffu)  // global_addr[56:32]
         | 0x80000000u;                        // type = 2 ("image")
  i32x8 g1;
  g1[0] = (int)(1u << 16);                     // data_size=1 -> 2 bytes
  g1[1] = (int)((ld & 0xffffu) << 16);         // tensor_dim0[15:0]
  g1[2] = (int)((ld >> 16) | ((nrows & 0xffffu) << 16)); // dim0 hi | dim1 lo
  g1[3] = (int)((nrows >> 16) | (32u << 16));  // dim1 hi | tile_dim0=32
  g1[4] = 64;                                  // tile_dim1=64, tile_dim2=0
  g1[5] = (int)ld;                             // tensor_dim0_stride[31:0]
  g1[6] = 0;                                   // stride hi | dim1_stride lo
  g1[7] = 0;
  i32x4 z4 = {};
#if defined(__clang_major__) && (__clang_major__ >= 23)
  i32x8 z8 = {};
  __builtin_amdgcn_tensor_load_to_lds(g0, g1, z4, z4, z8, 0);
#else
  __builtin_amdgcn_tensor_load_to_lds(g0, g1, z4, z4, 0);
#endif
}

// ---------------------------------------------------------------------------
// TDM-fed GEMM: C[M][N](f32) = A[M][K](f16) * W[N][K]^T (f16).
// Block = 4 waves -> 64x64 output tile. Per 32-wide K step the TDM streams a
// 64x32 A tile and a 64x32 B tile into double-buffered LDS; wave 0 issues the
// DMA and gates it with s_wait_tensorcnt; all math operands come from LDS.
// ---------------------------------------------------------------------------
__global__ void __launch_bounds__(128)
gemm_f16_tdm(const f16* __restrict__ A, const f16* __restrict__ W,
             float* __restrict__ C, int M, int N, int K) {
  __shared__ f16 sA[2][64][32];
  __shared__ f16 sB[2][64][32];

  int tid  = threadIdx.x;
  int lane = tid & 31;
  int wv   = tid >> 5;                  // wave 0..3
  int n0   = blockIdx.x * 64;
  int m0   = blockIdx.y * 64;
  bool issuer = (wv == 0);

  unsigned offA[2] = { (unsigned)(unsigned long long)&sA[0][0][0],
                       (unsigned)(unsigned long long)&sA[1][0][0] };
  unsigned offB[2] = { (unsigned)(unsigned long long)&sB[0][0][0],
                       (unsigned)(unsigned long long)&sB[1][0][0] };

  const f16* Abase = A + (size_t)m0 * K;
  const f16* Wbase = W + (size_t)n0 * K;

  if (issuer) {                         // prologue: pair for step 0 -> buf 0
    tdm_load_tile64x32(offA[0], Abase, (unsigned)K, 64u);
    tdm_load_tile64x32(offB[0], Wbase, (unsigned)K, 64u);
  }

  v8f acc[4] = {};
  int arow  = wv * 16 + (lane & 15);    // A row within 64-row tile
  int bcol  = lane & 15;                // B column within 16-wide subtile
  int khalf = (lane >> 4) << 4;         // B-frag K offset (0 or 16)
  int abase = (lane >> 4) << 4;         // A-frag K offset (0 or 16)

  int nsteps = K >> 5;
  for (int s = 0; s < nsteps; ++s) {
    int buf = s & 1;
    __syncthreads();                    // everyone done reading buf^1
    if (issuer) {
      if (s + 1 < nsteps) {
        int k1 = (s + 1) << 5;
        tdm_load_tile64x32(offA[buf ^ 1], Abase + k1, (unsigned)K, 64u);
        tdm_load_tile64x32(offB[buf ^ 1], Wbase + k1, (unsigned)K, 64u);
        __builtin_amdgcn_s_wait_tensorcnt(2);   // pair for step s complete
      } else {
        __builtin_amdgcn_s_wait_tensorcnt(0);
      }
    }
    __syncthreads();                    // tile for step s visible to all waves

    V16u ua;
    const f16* ap = &sA[buf][arow][abase];
    ua.h[0] = *(const v8h*)ap;
    ua.h[1] = *(const v8h*)(ap + 8);
    v16h a = ua.v;
#pragma unroll
    for (int nt = 0; nt < 4; ++nt) {
      v16h b = *(const v16h*)&sB[buf][nt * 16 + bcol][khalf];
      acc[nt] = wmma_f16(a, b, acc[nt]);
    }
  }

  int rbase = (lane >> 4) * 8;
#pragma unroll
  for (int nt = 0; nt < 4; ++nt)
#pragma unroll
    for (int r = 0; r < 8; ++r)
      C[(size_t)(m0 + wv * 16 + rbase + r) * N + n0 + nt * 16 + bcol] = acc[nt][r];
}
#endif  // USE_TDM

// ---------------------------------------------------------------------------
// Fallback GEMM (direct global loads), also used if TDM builtin is absent.
// ---------------------------------------------------------------------------
__global__ void __launch_bounds__(32)
gemm_f16(const f16* __restrict__ A, const f16* __restrict__ W,
         float* __restrict__ C, int M, int N, int K) {
  int lane = threadIdx.x;
  int n0 = blockIdx.x * 64;
  int m0 = blockIdx.y * 16;

  v8f acc[4] = {};
  int rowA  = m0 + (lane & 15);
  int bcol  = lane & 15;
  int khalf = (lane >> 4) << 4;

  for (int k0 = 0; k0 < K; k0 += 32) {
    v16h a = load_a_frag(A, K, rowA, k0, lane);
#pragma unroll
    for (int nt = 0; nt < 4; ++nt) {
      const f16* pb = W + (size_t)(n0 + nt * 16 + bcol) * K + k0 + khalf;
      v16h b = *(const v16h*)pb;
      acc[nt] = wmma_f16(a, b, acc[nt]);
    }
  }
  int rbase = (lane >> 4) * 8;
#pragma unroll
  for (int nt = 0; nt < 4; ++nt)
#pragma unroll
    for (int r = 0; r < 8; ++r)
      C[(size_t)(m0 + rbase + r) * N + n0 + nt * 16 + bcol] = acc[nt][r];
}

// ---------------------------------------------------------------------------
// RoPE + pack into attention-friendly f16 layouts.
//  Qh: [H][T][64]   (A-matrix layout source, scale 1/sqrt(64) folded in)
//  Kh: [Hkv][T][64] (B-frag: contiguous dims per key row)
//  Vt: [Hkv][64][T] (B-frag for P*V: contiguous keys per dim row)
// ---------------------------------------------------------------------------
__global__ void __launch_bounds__(896)
rope_pack(const float* __restrict__ Qf, const float* __restrict__ Kf,
          const float* __restrict__ Vf, f16* __restrict__ Qh,
          f16* __restrict__ Kh, f16* __restrict__ Vt) {
  int t = blockIdx.x;
  int tid = threadIdx.x;
  if (tid < 512) {                       // Q: 16 heads x 32 rotation pairs
    int h = tid >> 5, d = tid & 31;
    float inv = __powf(10000.0f, -(float)(2 * d) * (1.0f / 64.0f));
    float sn, cs;
    __sincosf((float)t * inv, &sn, &cs);
    const float* q = Qf + (size_t)t * C_EMB + h * 64;
    float x1 = q[d], x2 = q[d + 32];
    f16* o = Qh + (size_t)h * T_SEQ * 64 + (size_t)t * 64;
    o[d]      = (f16)((x1 * cs - x2 * sn) * 0.125f);
    o[d + 32] = (f16)((x2 * cs + x1 * sn) * 0.125f);
  } else if (tid < 640) {                // K: 4 heads x 32 pairs
    int u = tid - 512;
    int h = u >> 5, d = u & 31;
    float inv = __powf(10000.0f, -(float)(2 * d) * (1.0f / 64.0f));
    float sn, cs;
    __sincosf((float)t * inv, &sn, &cs);
    const float* k = Kf + (size_t)t * KV_DIM + h * 64;
    float x1 = k[d], x2 = k[d + 32];
    f16* o = Kh + (size_t)h * T_SEQ * 64 + (size_t)t * 64;
    o[d]      = (f16)(x1 * cs - x2 * sn);
    o[d + 32] = (f16)(x2 * cs + x1 * sn);
  } else if (tid < 896) {                // V transpose: 4 heads x 64 dims
    int u = tid - 640;
    int h = u >> 6, d = u & 63;
    Vt[(size_t)h * 64 * T_SEQ + (size_t)d * T_SEQ + t] =
        (f16)Vf[(size_t)t * KV_DIM + h * 64 + d];
  }
}

// ---------------------------------------------------------------------------
// Sliding-window flash attention. Grid: (T/16, N_HEAD), one wave per block.
// Per 32-key block: 4 wmma for S = Q*K^T, online softmax, P relayout C->A
// through LDS (same-wave DS ops are in-order), 4 wmma for O += P*V.
// Output Ob: [T][H*64] f16 (matches y.transpose(0,2,1,3).reshape).
// ---------------------------------------------------------------------------
__global__ void __launch_bounds__(32)
attn_swa(const f16* __restrict__ Qh, const f16* __restrict__ Kh,
         const f16* __restrict__ Vt, f16* __restrict__ Ob) {
  __shared__ f16 sP[16][32];

  int lane = threadIdx.x;
  int q0 = blockIdx.x * 16;
  int h  = blockIdx.y;
  int kvh = h >> 2;                       // GQA: 16 heads over 4 KV heads

  const f16* Q = Qh + (size_t)h   * T_SEQ * 64;
  const f16* K = Kh + (size_t)kvh * T_SEQ * 64;
  const f16* V = Vt + (size_t)kvh * 64 * T_SEQ;

  int qrow = q0 + (lane & 15);
  v16h qa0 = load_a_frag(Q, 64, qrow, 0,  lane);
  v16h qa1 = load_a_frag(Q, 64, qrow, 32, lane);

  v8f accO[4] = {};
  float m[8], ssum[8];
#pragma unroll
  for (int r = 0; r < 8; ++r) { m[r] = -1e30f; ssum[r] = 0.0f; }

  int rowoff = (lane >> 4) * 8;           // C/D: rows r (lanes<16) or r+8
  int coln   = lane & 15;
  int khalf  = (lane >> 4) << 4;

  int lo = q0 - (WIN - 1);
  if (lo < 0) lo = 0;
  int jb = lo & ~31;

  for (int j0 = jb; j0 <= q0 + 15; j0 += 32) {
    // ---- S = Q * K^T for keys [j0, j0+32) ----
    v8f s0 = {}, s1 = {};
    {
      const f16* k0p = K + (size_t)(j0 + coln) * 64 + khalf;
      const f16* k1p = K + (size_t)(j0 + 16 + coln) * 64 + khalf;
      s0 = wmma_f16(qa0, *(const v16h*)(k0p),      s0);
      s1 = wmma_f16(qa0, *(const v16h*)(k1p),      s1);
      s0 = wmma_f16(qa1, *(const v16h*)(k0p + 32), s0);
      s1 = wmma_f16(qa1, *(const v16h*)(k1p + 32), s1);
    }

    // ---- masked online softmax ----
    int j_0 = j0 + coln, j_1 = j0 + 16 + coln;
#pragma unroll
    for (int r = 0; r < 8; ++r) {
      int i = q0 + rowoff + r;
      float v0 = ((j_0 <= i) && (j_0 > i - WIN)) ? s0[r] : -1e30f;
      float v1 = ((j_1 <= i) && (j_1 > i - WIN)) ? s1[r] : -1e30f;
      float bm = fmaxf(v0, v1);
#pragma unroll
      for (int off = 8; off >= 1; off >>= 1)
        bm = fmaxf(bm, __shfl_xor(bm, off, 32));   // stays within 16-lane half
      float mn = fmaxf(m[r], bm);
      float alpha = __expf(m[r] - mn);             // underflows to 0 on first real block
      float p0 = __expf(v0 - mn);
      float p1 = __expf(v1 - mn);
      float rs = p0 + p1;
#pragma unroll
      for (int off = 8; off >= 1; off >>= 1)
        rs += __shfl_xor(rs, off, 32);
      ssum[r] = ssum[r] * alpha + rs;
      m[r] = mn;
#pragma unroll
      for (int nt = 0; nt < 4; ++nt) accO[nt][r] *= alpha;
      sP[rowoff + r][coln]      = (f16)p0;
      sP[rowoff + r][16 + coln] = (f16)p1;
    }

    // ---- P: C-layout -> A-layout via LDS (same-wave DS ordering) ----
    V16u u;
    int prow = lane & 15;
    int pb = (lane >> 4) * 16;
    u.h[0] = *(const v8h*)&sP[prow][pb];
    u.h[1] = *(const v8h*)&sP[prow][pb + 8];
    v16h pf = u.v;

    // ---- O += P * V ----
#pragma unroll
    for (int nt = 0; nt < 4; ++nt) {
      const f16* vp = V + (size_t)(nt * 16 + coln) * T_SEQ + j0 + khalf;
      accO[nt] = wmma_f16(pf, *(const v16h*)vp, accO[nt]);
    }
  }

  // ---- normalize and store [T][H*64] ----
#pragma unroll
  for (int nt = 0; nt < 4; ++nt)
#pragma unroll
    for (int r = 0; r < 8; ++r) {
      float o = accO[nt][r] / ssum[r];
      Ob[(size_t)(q0 + rowoff + r) * C_EMB + h * 64 + nt * 16 + coln] = (f16)o;
    }
}

// ---------------------------------------------------------------------------
// Launch
// ---------------------------------------------------------------------------
static inline void launch_gemm(const f16* A, const f16* W, float* C,
                               int M, int N, int K, hipStream_t stream) {
#if USE_TDM
  gemm_f16_tdm<<<dim3(N / 64, M / 64), 128, 0, stream>>>(A, W, C, M, N, K);
#else
  gemm_f16<<<dim3(N / 64, M / 16), 32, 0, stream>>>(A, W, C, M, N, K);
#endif
}

extern "C" void kernel_launch(void* const* d_in, const int* in_sizes, int n_in,
                              void* d_out, int out_size, void* d_ws, size_t ws_size,
                              hipStream_t stream) {
  const float* x  = (const float*)d_in[0];   // [4096][1024]
  const float* wq = (const float*)d_in[1];   // [1024][1024]
  const float* wk = (const float*)d_in[2];   // [256][1024]
  const float* wv = (const float*)d_in[3];   // [256][1024]
  const float* wo = (const float*)d_in[4];   // [1024][1024]
  float* out = (float*)d_out;                // [4096][1024]

  char* ws = (char*)d_ws;
  // f16 buffers
  f16* xb  = (f16*)(ws + 0);                              //  8 MiB
  f16* wqh = (f16*)(ws + 8388608);                        //  2 MiB
  f16* wkh = (f16*)(ws + 10485760);                       //  0.5 MiB
  f16* wvh = (f16*)(ws + 11010048);                       //  0.5 MiB
  f16* woh = (f16*)(ws + 11534336);                       //  2 MiB
  // fp32 projection outputs
  float* Qf = (float*)(ws + 13631488);                    // 16 MiB
  float* Kf = (float*)(ws + 30408704);                    //  4 MiB
  float* Vf = (float*)(ws + 34603008);                    //  4 MiB
  // packed f16 attention operands
  f16* Qhh = (f16*)(ws + 38797312);                       //  8 MiB
  f16* Khh = (f16*)(ws + 47185920);                       //  2 MiB
  f16* Vtt = (f16*)(ws + 49283072);                       //  2 MiB  (end ~49 MiB)
  // attention output aliases the dead Qf region (f16, 8 MiB)
  f16* Obb = (f16*)(ws + 13631488);

  // 1) casts
  cast_f32_to_f16<<<(T_SEQ * C_EMB + 255) / 256, 256, 0, stream>>>(x,  xb,  T_SEQ * C_EMB);
  cast_f32_to_f16<<<(C_EMB * C_EMB + 255) / 256, 256, 0, stream>>>(wq, wqh, C_EMB * C_EMB);
  cast_f32_to_f16<<<(KV_DIM * C_EMB + 255) / 256, 256, 0, stream>>>(wk, wkh, KV_DIM * C_EMB);
  cast_f32_to_f16<<<(KV_DIM * C_EMB + 255) / 256, 256, 0, stream>>>(wv, wvh, KV_DIM * C_EMB);
  cast_f32_to_f16<<<(C_EMB * C_EMB + 255) / 256, 256, 0, stream>>>(wo, woh, C_EMB * C_EMB);

  // 2) QKV projections (WMMA, TDM-fed)
  launch_gemm(xb, wqh, Qf, T_SEQ, C_EMB,  C_EMB, stream);
  launch_gemm(xb, wkh, Kf, T_SEQ, KV_DIM, C_EMB, stream);
  launch_gemm(xb, wvh, Vf, T_SEQ, KV_DIM, C_EMB, stream);

  // 3) RoPE + pack
  rope_pack<<<T_SEQ, 896, 0, stream>>>(Qf, Kf, Vf, Qhh, Khh, Vtt);

  // 4) sliding-window attention (WMMA flash)
  attn_swa<<<dim3(T_SEQ / 16, N_HEAD), 32, 0, stream>>>(Qhh, Khh, Vtt, Obb);

  // 5) output projection (WMMA) -> fp32 out
  launch_gemm(Obb, woh, out, T_SEQ, C_EMB, C_EMB, stream);
}
